// FactorizationMachineLayer_65712999629187
// MI455X (gfx1250) — matrix operand order repaired and checked by reading, not compiled, not run on hardware
//
#include <hip/hip_runtime.h>
#include <hip/hip_bf16.h>

typedef __attribute__((ext_vector_type(16))) _Float16 v16h;
typedef __attribute__((ext_vector_type(8)))  float    v8f;
typedef int v4i __attribute__((vector_size(16)));
typedef __attribute__((address_space(1))) v4i* as1_v4i_ptr;
typedef __attribute__((address_space(3))) v4i* as3_v4i_ptr;

#define FM_B        4096
#define FM_VOCAB    50000
#define FM_F        64
#define FM_VPAD     50048                 // VOCAB padded (covers B-prefetch overshoot)
#define FM_KSTEPS   1563                  // ceil(50000 / 32)
#define FM_KCHUNKS  16
#define FM_SPC      98                    // ceil(1563 / 16) steps per K-chunk

#define FM_DEPTH    3                     // async stages in flight ahead of consumer
#define FM_NBUF     4                     // ring slots (DEPTH+1: no LDS WAR race)

// workspace layout (bytes); d_ws is 256B-aligned from hipMalloc
#define WS_OFF_VH   0u
#define WS_OFF_C    (FM_VPAD * FM_F * 2u)             // Vh: 6,406,144 B
#define WS_OFF_XV   (WS_OFF_C + FM_VPAD * 4u)         // c : 200,192 B
#define WS_OFF_LIN  (WS_OFF_XV + FM_B * FM_F * 4u)    // XV: 1,048,576 B ; lin: 16,384 B
// total ~7.7 MB

// ---------------------------------------------------------------------------
// prep: Vh = f16(V) (zero-padded rows), c[i] = bias[i] - 0.5*||V_i||^2 (padded 0)
// ---------------------------------------------------------------------------
__global__ void fm_prep(const float* __restrict__ V, const float* __restrict__ bias,
                        _Float16* __restrict__ Vh, float* __restrict__ c) {
    const int i = blockIdx.x * blockDim.x + threadIdx.x;
    if (i >= FM_VPAD) return;
    _Float16* hr = Vh + (size_t)i * FM_F;
    if (i < FM_VOCAB) {
        const float* vr = V + (size_t)i * FM_F;
        float ss = 0.f;
#pragma unroll
        for (int j = 0; j < FM_F; j += 4) {
            float4 v = *(const float4*)(vr + j);
            ss += v.x * v.x + v.y * v.y + v.z * v.z + v.w * v.w;
            hr[j + 0] = (_Float16)v.x;
            hr[j + 1] = (_Float16)v.y;
            hr[j + 2] = (_Float16)v.z;
            hr[j + 3] = (_Float16)v.w;
        }
        c[i] = bias[i] - 0.5f * ss;
    } else {
#pragma unroll
        for (int j = 0; j < FM_F; ++j) hr[j] = (_Float16)0.0f;
        c[i] = 0.f;
    }
}

__global__ void fm_zero(float* __restrict__ p, int n) {
    const int i = blockIdx.x * blockDim.x + threadIdx.x;
    if (i < n) p[i] = 0.f;
}

// ---------------------------------------------------------------------------
// async copy helpers (gfx1250 GLOBAL_LOAD_ASYNC_TO_LDS_B128, ASYNCcnt tracked)
// ---------------------------------------------------------------------------
__device__ __forceinline__ void fm_async_b128(const int* gsrc, int* ldst) {
    __builtin_amdgcn_global_load_async_to_lds_b128(
        (as1_v4i_ptr)(uintptr_t)gsrc,
        (as3_v4i_ptr)(uint32_t)(uintptr_t)ldst,
        0, 0);
}

// Stage one 16-row x 32-int x tile (2 KB) into this wave's ring slot.
// chunk c = j*32+lane : row = c>>3, intoff = (c&7)*4 ; LDS byte off = c*16.
__device__ __forceinline__ void fm_issue_stage(const int* __restrict__ x, int m0,
                                               int lane, int* __restrict__ slotbuf,
                                               int k0) {
#pragma unroll
    for (int j = 0; j < 4; ++j) {
        const int c   = j * 32 + lane;
        const int row = c >> 3;
        const int off = (c & 7) * 4;
        int kk = k0 + off;
        if (kk > FM_VOCAB - 4) kk = FM_VOCAB - 4;   // stay in-bounds; garbage*0 = 0
        const int* g = x + (size_t)(m0 + row) * FM_VOCAB + kk;
        fm_async_b128(g, slotbuf + c * 4);
    }
}

// ---------------------------------------------------------------------------
// main: K-split WMMA GEMM; x streamed HBM->LDS asynchronously (3 stages deep),
// B (f16 V rows + c column) register-double-buffered. 5 WMMA / K-step.
// ---------------------------------------------------------------------------
#define XH(v) (((v) != 0) ? (_Float16)1.0f : (_Float16)0.0f)

__global__ __launch_bounds__(128) void fm_main(const int* __restrict__ x,
                                               const _Float16* __restrict__ Vh,
                                               const float* __restrict__ cbuf,
                                               float* __restrict__ XV,
                                               float* __restrict__ lin) {
    __shared__ int ldsx[4][FM_NBUF][512];            // 4 waves * 4 slots * 2 KB = 32 KB

    const int lane = threadIdx.x & 31;
    const int wave = threadIdx.x >> 5;
    const int m0   = (blockIdx.x * 4 + wave) * 16;
    const int sBeg = blockIdx.y * FM_SPC;
    const int sEnd = (sBeg + FM_SPC < FM_KSTEPS) ? (sBeg + FM_SPC) : FM_KSTEPS;

    // A-matrix 16x32 f16 layout: lanes 0-15 -> M=lane, K halves {0..7,16..23};
    // lanes 16-31 -> M=lane-16, K halves {8..15,24..31}.
    const int row  = lane & 15;
    const int koff = (lane & 16) ? 8 : 0;

    v8f acc0 = {}, acc1 = {}, acc2 = {}, acc3 = {}, accL = {};

    // ---- prologue: launch FM_DEPTH x-stages, load first B tiles ----
#pragma unroll
    for (int d = 0; d < FM_DEPTH; ++d)
        if (sBeg + d < sEnd)
            fm_issue_stage(x, m0, lane, ldsx[wave][(sBeg + d) & (FM_NBUF - 1)],
                           (sBeg + d) * 32);

    const _Float16* vrow0 = Vh + (size_t)(sBeg * 32) * FM_F + row;
    v16h b0, b1, b2, b3;
    v16h b5 = {};
#pragma unroll
    for (int e = 0; e < 16; ++e) {
        const int kk = (e < 8) ? (koff + e) : (16 + koff + e - 8);
        b0[e] = vrow0[kk * FM_F +  0];
        b1[e] = vrow0[kk * FM_F + 16];
        b2[e] = vrow0[kk * FM_F + 32];
        b3[e] = vrow0[kk * FM_F + 48];
        if (row == 0) b5[e] = (_Float16)cbuf[sBeg * 32 + kk];
    }

    for (int s = sBeg; s < sEnd; ++s) {
        // retire the oldest stage: <= (DEPTH-1)*4 ops still outstanding
        __builtin_amdgcn_s_wait_asynccnt(8);

        // ---- A tile from LDS ring slot (ds_load_b128 x4), int -> f16 {0,1} ----
        const int* lrow = &ldsx[wave][s & (FM_NBUF - 1)][row * 32];
        int4 u0 = *(const int4*)(lrow + koff);
        int4 u1 = *(const int4*)(lrow + koff + 4);
        int4 u2 = *(const int4*)(lrow + 16 + koff);
        int4 u3 = *(const int4*)(lrow + 20 + koff);
        v16h a;
        a[0]  = XH(u0.x); a[1]  = XH(u0.y); a[2]  = XH(u0.z); a[3]  = XH(u0.w);
        a[4]  = XH(u1.x); a[5]  = XH(u1.y); a[6]  = XH(u1.z); a[7]  = XH(u1.w);
        a[8]  = XH(u2.x); a[9]  = XH(u2.y); a[10] = XH(u2.z); a[11] = XH(u2.w);
        a[12] = XH(u3.x); a[13] = XH(u3.y); a[14] = XH(u3.z); a[15] = XH(u3.w);

        // ---- keep the async pipe FM_DEPTH stages deep ----
        if (s + FM_DEPTH < sEnd)
            fm_issue_stage(x, m0, lane, ldsx[wave][(s + FM_DEPTH) & (FM_NBUF - 1)],
                           (s + FM_DEPTH) * 32);

        // ---- B double buffer: fetch next step's tiles before this step's WMMAs
        //      (Vh/c are zero-padded past VOCAB, so overshoot contributes 0) ----
        const _Float16* vrow = Vh + (size_t)((s + 1) * 32) * FM_F + row;
        v16h nb0, nb1, nb2, nb3;
        v16h nb5 = {};
#pragma unroll
        for (int e = 0; e < 16; ++e) {
            const int kk = (e < 8) ? (koff + e) : (16 + koff + e - 8);
            nb0[e] = vrow[kk * FM_F +  0];
            nb1[e] = vrow[kk * FM_F + 16];
            nb2[e] = vrow[kk * FM_F + 32];
            nb3[e] = vrow[kk * FM_F + 48];
            if (row == 0) nb5[e] = (_Float16)cbuf[(s + 1) * 32 + kk];
        }

        acc0 = __builtin_amdgcn_wmma_f32_16x16x32_f16(false, a, false, b0, (short)0, acc0, false, false);
        acc1 = __builtin_amdgcn_wmma_f32_16x16x32_f16(false, a, false, b1, (short)0, acc1, false, false);
        acc2 = __builtin_amdgcn_wmma_f32_16x16x32_f16(false, a, false, b2, (short)0, acc2, false, false);
        acc3 = __builtin_amdgcn_wmma_f32_16x16x32_f16(false, a, false, b3, (short)0, acc3, false, false);
        accL = __builtin_amdgcn_wmma_f32_16x16x32_f16(false, a, false, b5, (short)0, accL, false, false);

        b0 = nb0; b1 = nb1; b2 = nb2; b3 = nb3; b5 = nb5;
    }

    // ---- reduce K-partials into global XV / lin ----
    // C layout: VGPR r: lanes 0-15 -> (M=r, N=lane), lanes 16-31 -> (M=r+8, N=lane-16)
    const int l   = lane & 15;
    const int mhi = (lane >> 4) << 3;                // 0 or 8
    float* base = XV + (size_t)(m0 + mhi) * FM_F + l;
#pragma unroll
    for (int r = 0; r < 8; ++r) {
        atomicAdd(base + (size_t)r * FM_F +  0, acc0[r]);
        atomicAdd(base + (size_t)r * FM_F + 16, acc1[r]);
        atomicAdd(base + (size_t)r * FM_F + 32, acc2[r]);
        atomicAdd(base + (size_t)r * FM_F + 48, acc3[r]);
    }
    if (l == 0) {                                    // column 0 of the c-tile
#pragma unroll
        for (int r = 0; r < 8; ++r)
            atomicAdd(&lin[m0 + mhi + r], accL[r]);
    }
}

// ---------------------------------------------------------------------------
// finish: out[m] = g_bias + lin[m] + 0.5 * ||XV[m,:]||^2
// ---------------------------------------------------------------------------
__global__ void fm_finish(const float* __restrict__ XV, const float* __restrict__ lin,
                          const float* __restrict__ g_bias, float* __restrict__ out) {
    const int m = blockIdx.x * blockDim.x + threadIdx.x;
    if (m >= FM_B) return;
    const float* xv = XV + (size_t)m * FM_F;
    float s = 0.f;
#pragma unroll
    for (int j = 0; j < FM_F; j += 4) {
        float4 v = *(const float4*)(xv + j);
        s += v.x * v.x + v.y * v.y + v.z * v.z + v.w * v.w;
    }
    out[m] = g_bias[0] + lin[m] + 0.5f * s;
}

extern "C" void kernel_launch(void* const* d_in, const int* in_sizes, int n_in,
                              void* d_out, int out_size, void* d_ws, size_t ws_size,
                              hipStream_t stream) {
    const int*   x      = (const int*)d_in[0];      // [B, VOCAB] int32 {0,1}
    const float* V      = (const float*)d_in[1];    // [VOCAB, F]
    const float* bias   = (const float*)d_in[2];    // [VOCAB, 1]
    const float* g_bias = (const float*)d_in[3];    // scalar
    float*       out    = (float*)d_out;            // [B, 1]

    char* ws = (char*)d_ws;
    _Float16* Vh  = (_Float16*)(ws + WS_OFF_VH);
    float*    c   = (float*)(ws + WS_OFF_C);
    float*    XV  = (float*)(ws + WS_OFF_XV);
    float*    lin = (float*)(ws + WS_OFF_LIN);

    (void)in_sizes; (void)n_in; (void)out_size; (void)ws_size;

    fm_prep<<<(FM_VPAD + 255) / 256, 256, 0, stream>>>(V, bias, Vh, c);

    const int nz = FM_B * FM_F + FM_B;               // XV ++ lin contiguous
    fm_zero<<<(nz + 255) / 256, 256, 0, stream>>>(XV, nz);

    dim3 grid(FM_B / 64, FM_KCHUNKS);                // 4 waves/block * 16 rows = 64 rows
    fm_main<<<grid, 128, 0, stream>>>(x, Vh, c, XV, lin);

    fm_finish<<<(FM_B + 255) / 256, 256, 0, stream>>>(XV, lin, g_bias, out);
}
